// PixProLoss_6279242187477
// MI455X (gfx1250) — matrix operand order repaired and checked by reading, not compiled, not run on hardware
//
#include <hip/hip_runtime.h>
#include <hip/hip_bf16.h>

typedef __attribute__((ext_vector_type(16))) _Float16 v16h;
typedef __attribute__((ext_vector_type(8)))  float    v8f;

#define NB   8
#define CH   64
#define HW   3136           // 56*56
#define TILES 196           // HW / 16
#define TJ_PER_WAVE 4
#define JGROUPS (TILES / TJ_PER_WAVE)          // 49
#define GROUPS_PER_BATCH (TILES * JGROUPS)     // 9604
#define WAVES_PER_BLOCK 4
#define BLOCKS_PER_BATCH (GROUPS_PER_BATCH / WAVES_PER_BLOCK)  // 2401, exact
#define CHUNK_HALFS 512     // 32 lanes * 16 halfs per K-chunk fragment
#define QA_BYTES ((size_t)NB * TILES * 2 * CHUNK_HALFS * 2)    // 3,211,264 B

// ---------------------------------------------------------------------------
// Kernel 0: zero the per-batch accumulators (sumLogit, sumMask) x 8 batches.
// ---------------------------------------------------------------------------
__global__ void pixpro_init(float* acc) {
    if (threadIdx.x < 16) acc[threadIdx.x] = 0.0f;
}

// ---------------------------------------------------------------------------
// Kernel 1: L2-normalize q,k over C per pixel; scatter f16 into WMMA fragment
// layouts (ISA 7.12.2, wave32 16-bit A/B layouts).
//   A fragment (16x32 f16, M x K), lane L: M = L&15, kb = (L>=16)?8:0
//     element e (0..15): K = (e>>3)*16 + kb + (e&7)
//   B fragment (32x16 f16, K x N), lane L: N = L&15, kb = (L>=16)?8:0
//     element e (0..15): K = (e>>3)*16 + kb + (e&7)
// Packed as [n][tile][chunk(0..1)][lane(0..31)][elem(0..15)].
// ---------------------------------------------------------------------------
__global__ void normalize_pack(const float* __restrict__ q,
                               const float* __restrict__ k,
                               _Float16* __restrict__ Qa,
                               _Float16* __restrict__ Kb) {
    int idx = blockIdx.x * blockDim.x + threadIdx.x;   // over N*HW = 25088
    if (idx >= NB * HW) return;
    int n = idx / HW;
    int p = idx - n * HW;

    // norms over C (stride-HW reads are coalesced across adjacent lanes)
    float sq = 0.f, sk = 0.f;
    #pragma unroll 8
    for (int c = 0; c < CH; ++c) {
        float vq = q[((size_t)n * CH + c) * HW + p];
        float vk = k[((size_t)n * CH + c) * HW + p];
        sq += vq * vq;
        sk += vk * vk;
    }
    float invq = 1.0f / fmaxf(sqrtf(sq), 1e-8f);
    float invk = 1.0f / fmaxf(sqrtf(sk), 1e-8f);

    int tile = p >> 4;
    int m    = p & 15;      // row within tile (A) / col within tile (B)
    size_t base = (((size_t)n * TILES + tile) * 2) * CHUNK_HALFS;

    #pragma unroll 8
    for (int c = 0; c < CH; ++c) {
        int chunk = c >> 5;
        int ck    = c & 31;
        float vq = q[((size_t)n * CH + c) * HW + p] * invq;
        float vk = k[((size_t)n * CH + c) * HW + p] * invk;

        // ---- A-fragment scatter (Q) ----
        int g  = ck >> 4;          // 0 or 1 (K 0..15 vs 16..31)
        int r  = ck & 15;
        int laneA = m + ((r & 8) ? 16 : 0);
        int eA    = g * 8 + (r & 7);
        Qa[base + (size_t)chunk * CHUNK_HALFS + laneA * 16 + eA] = (_Float16)vq;

        // ---- B-fragment scatter (K) ----
        int laneB = m + ((ck & 8) ? 16 : 0);
        int eB    = (ck >> 4) * 8 + (ck & 7);
        Kb[base + (size_t)chunk * CHUNK_HALFS + laneB * 16 + eB] = (_Float16)vk;
    }
}

// ---------------------------------------------------------------------------
// Kernel 2: each wave computes a 16 x 64 strip (one i-tile x 4 j-tiles).
// A fragments + the 8 row centers are loaded once; per j-tile we load B,
// issue 2 chained WMMAs (K=64), mask with dist^2 < 0.49 and accumulate.
// One wave shuffle-reduction + LDS block reduction + 2 atomics per block.
// ---------------------------------------------------------------------------
__global__ __launch_bounds__(128)
void masked_gemm(const _Float16* __restrict__ Qa,
                 const _Float16* __restrict__ Kb,
                 const float* __restrict__ cqx, const float* __restrict__ cqy,
                 const float* __restrict__ ckx, const float* __restrict__ cky,
                 float* __restrict__ acc) {
    int wave = threadIdx.x >> 5;
    int lane = threadIdx.x & 31;
    int n    = blockIdx.y;
    int grp  = blockIdx.x * WAVES_PER_BLOCK + wave;   // always < GROUPS_PER_BATCH
    int ti = grp / JGROUPS;
    int tg = grp - ti * JGROUPS;
    int tj0 = tg * TJ_PER_WAVE;

    // ---- A fragments: loaded once, reused for all 4 j-tiles ----
    const v16h* A = (const v16h*)(Qa + (((size_t)n * TILES + ti) * 2) * CHUNK_HALFS);
    v16h a0 = A[lane];          // K chunk 0..31
    v16h a1 = A[32 + lane];     // K chunk 32..63

    // ---- row centers for this lane's 8 D-rows: loaded once ----
    int nn = lane & 15;
    int hi = lane >> 4;
    int i0 = ti * 16 + hi * 8;  // D layout: VGPR v -> M = v + 8*(lane>=16)
    size_t nb = (size_t)n * HW;
    float qx[8], qy[8];
    #pragma unroll
    for (int v = 0; v < 8; ++v) {
        qx[v] = cqx[nb + i0 + v];
        qy[v] = cqy[nb + i0 + v];
    }

    float sL = 0.f, sM = 0.f;

    #pragma unroll
    for (int jj = 0; jj < TJ_PER_WAVE; ++jj) {
        int tj = tj0 + jj;
        const v16h* B = (const v16h*)(Kb + (((size_t)n * TILES + tj) * 2) * CHUNK_HALFS);
        v16h b0 = B[lane];
        v16h b1 = B[32 + lane];

        v8f c = {};
        c = __builtin_amdgcn_wmma_f32_16x16x32_f16(false, a0, false, b0,
                                                   (short)0, c, false, false);
        c = __builtin_amdgcn_wmma_f32_16x16x32_f16(false, a1, false, b1,
                                                   (short)0, c, false, false);

        int j = tj * 16 + nn;   // D layout: lane -> column N
        float kx = ckx[nb + j];
        float ky = cky[nb + j];

        #pragma unroll
        for (int v = 0; v < 8; ++v) {
            float dx = qx[v] - kx;
            float dy = qy[v] - ky;
            bool m = (dx * dx + dy * dy) < 0.49f;   // dist < 0.7
            sL += m ? c[v] : 0.0f;
            sM += m ? 1.0f : 0.0f;
        }
    }

    // wave32 reduction (once per 16x64 strip)
    #pragma unroll
    for (int off = 16; off > 0; off >>= 1) {
        sL += __shfl_xor(sL, off, 32);
        sM += __shfl_xor(sM, off, 32);
    }

    __shared__ float lsL[WAVES_PER_BLOCK];
    __shared__ float lsM[WAVES_PER_BLOCK];
    if (lane == 0) { lsL[wave] = sL; lsM[wave] = sM; }
    __syncthreads();

    if (threadIdx.x == 0) {
        float tL = 0.f, tM = 0.f;
        #pragma unroll
        for (int w = 0; w < WAVES_PER_BLOCK; ++w) { tL += lsL[w]; tM += lsM[w]; }
        atomicAdd(&acc[n * 2 + 0], tL);
        atomicAdd(&acc[n * 2 + 1], tM);
    }
}

// ---------------------------------------------------------------------------
// Kernel 3: scalar finalize:  -2 * mean_n( sumLogit_n / (sumMask_n + 1e-6) )
// ---------------------------------------------------------------------------
__global__ void pixpro_finalize(const float* __restrict__ acc,
                                float* __restrict__ out) {
    if (threadIdx.x == 0 && blockIdx.x == 0) {
        float s = 0.f;
        #pragma unroll
        for (int n = 0; n < NB; ++n)
            s += acc[n * 2 + 0] / (acc[n * 2 + 1] + 1e-6f);
        out[0] = -2.0f * (s / (float)NB);
    }
}

extern "C" void kernel_launch(void* const* d_in, const int* in_sizes, int n_in,
                              void* d_out, int out_size, void* d_ws, size_t ws_size,
                              hipStream_t stream) {
    const float* q   = (const float*)d_in[0];
    const float* k   = (const float*)d_in[1];
    const float* cqx = (const float*)d_in[2];
    const float* cqy = (const float*)d_in[3];
    const float* ckx = (const float*)d_in[4];
    const float* cky = (const float*)d_in[5];
    float* out = (float*)d_out;

    char* ws = (char*)d_ws;
    _Float16* Qa  = (_Float16*)(ws);
    _Float16* Kb  = (_Float16*)(ws + QA_BYTES);
    float*    acc = (float*)(ws + 2 * QA_BYTES);

    pixpro_init<<<1, 32, 0, stream>>>(acc);

    int pix = NB * HW;                           // 25088
    normalize_pack<<<(pix + 255) / 256, 256, 0, stream>>>(q, k, Qa, Kb);

    dim3 grid(BLOCKS_PER_BATCH, NB);             // 2401 x 8
    masked_gemm<<<grid, 128, 0, stream>>>(Qa, Kb, cqx, cqy, ckx, cky, acc);

    pixpro_finalize<<<1, 1, 0, stream>>>(acc, out);
}
